// SuperPointNet_61117384622716
// MI455X (gfx1250) — compile-verified, matrix operand order checked by reading, not run on hardware
//
#include <hip/hip_runtime.h>

// ---------------------------------------------------------------------------
// SuperPoint-style net on MI455X (gfx1250, wave32, WMMA, TDM).
// Deform conv == fixed-gather im2col (offsets depend only on H,W) + f16 WMMA GEMM.
// Activations live in [C][B*H*W] f32; patches/weights in WMMA fragment layout (f16).
// GEMM: A (weight) panel staged into LDS by the Tensor Data Mover, B fragments
// software-pipelined from global.
// ---------------------------------------------------------------------------

typedef __attribute__((ext_vector_type(16))) _Float16     v16h;
typedef __attribute__((ext_vector_type(8)))  float        v8f;
typedef __attribute__((ext_vector_type(4)))  unsigned int v4u;
typedef __attribute__((ext_vector_type(8)))  int          v8i;
typedef __attribute__((ext_vector_type(4)))  int          v4i;

struct Tap { float w[4]; int i[4]; };   // 4 bilinear corners: weight (0 if OOB) + clipped index

static const long long PATCH_ELEMS = 1152LL * 16384LL;  // max fragment buffer (f16 elems)

// --- CDNA5 16-bit fragment maps (ISA 7.12.2) -------------------------------
// A 16x32: lane l holds row m=l%16 (half h=l>>4); v16h elem e -> VGPR j=e/2 half p=e&1,
//          K = (j<4 ? 2j+p+8h : 16+2(j-4)+p+8h).  B 32x16 is the mirror with col n=l%16.
__device__ __forceinline__ int fragK(int lane, int e) {
  int h = lane >> 4;
  int j = e >> 1, p = e & 1;
  return (j < 4) ? (2 * j + p + 8 * h) : (16 + 2 * (j - 4) + p + 8 * h);
}
__device__ __forceinline__ void fragPos(int kr, int r, int* lane, int* e) {
  int h  = (kr >> 3) & 1;
  int rr = kr & 7;
  int j  = ((kr & 16) ? 4 : 0) + (rr >> 1);
  *e    = 2 * j + (rr & 1);
  *lane = r + 16 * h;
}

// --- bilinear tap tables (per resolution, batch-invariant) -----------------
__global__ void build_tables_kernel(const float* __restrict__ coords,
                                    Tap* __restrict__ tab, int H, int W) {
  int tid = blockIdx.x * blockDim.x + threadIdx.x;
  int HW = H * W;
  if (tid >= 9 * HW) return;
  int k = tid / HW, p = tid - k * HW;
  int hy = p / W, wx = p - hy * W;
  float dh = coords[(2 * k) * HW + p];        // batch 0; offsets identical over batch
  float dw = coords[(2 * k + 1) * HW + p];
  float ys = (float)hy + (float)(k / 3 - 1) + dh;   // 3x3, pad 1
  float xs = (float)wx + (float)(k % 3 - 1) + dw;
  float y0 = floorf(ys), x0 = floorf(xs);
  float wy = ys - y0, wxx = xs - x0;
  int iy0 = (int)y0, ix0 = (int)x0;
  float wt[4] = {(1.f - wy) * (1.f - wxx), (1.f - wy) * wxx,
                 wy * (1.f - wxx),          wy * wxx};
  Tap t;
#pragma unroll
  for (int c = 0; c < 4; ++c) {
    int yy = iy0 + (c >> 1), xx = ix0 + (c & 1);
    bool valid = (yy >= 0) && (yy < H) && (xx >= 0) && (xx < W);
    int yc = min(max(yy, 0), H - 1), xc = min(max(xx, 0), W - 1);
    t.w[c] = valid ? wt[c] : 0.f;
    t.i[c] = yc * W + xc;
  }
  tab[tid] = t;
}

// --- weights -> A fragments (zero-padded) ----------------------------------
__global__ void pack_weights_kernel(const float* __restrict__ w,
                                    _Float16* __restrict__ frag,
                                    int M, int CK, int KB, int total) {
  int tid = blockIdx.x * blockDim.x + threadIdx.x;
  if (tid >= total) return;
  int e = tid & 15, lane = (tid >> 4) & 31;
  int idx = tid >> 9;
  int kb = idx % KB, mb = idx / KB;
  int m  = mb * 16 + (lane & 15);
  int kk = kb * 32 + fragK(lane, e);
  float v = (m < M && kk < CK) ? w[(size_t)m * CK + kk] : 0.f;
  frag[tid] = (_Float16)v;
}

// --- fused bilinear gather + swizzle into B fragments ----------------------
// grid: (nLen/256, Kpad); ck = blockIdx.y is block-uniform -> scalar /9.
__global__ void gather_patches_kernel(const float* __restrict__ act,
                                      const Tap* __restrict__ tab,
                                      _Float16* __restrict__ patches,
                                      int CK, int nStart, int nLen,
                                      int Ntot, int hwShift) {
  int nl = blockIdx.x * blockDim.x + threadIdx.x;
  int ck = blockIdx.y;
  float v = 0.f;
  if (ck < CK) {
    int c = ck / 9, k = ck - c * 9;          // uniform -> SALU
    int n = nStart + nl;
    int b = n >> hwShift, p = n & ((1 << hwShift) - 1);
    Tap t = tab[(k << hwShift) + p];
    const float* a = act + (size_t)c * Ntot + ((size_t)b << hwShift);
    v = t.w[0] * a[t.i[0]] + t.w[1] * a[t.i[1]] +
        t.w[2] * a[t.i[2]] + t.w[3] * a[t.i[3]];
  }
  int kb = ck >> 5, kr = ck & 31;
  int nb = nl >> 4, nr = nl & 15;
  int lane, e; fragPos(kr, nr, &lane, &e);
  int NB = nLen >> 4;
  patches[((size_t)(kb * NB + nb) * 32 + lane) * 16 + e] = (_Float16)v;
}

// --- 1x1 conv: activation -> B fragments (identity gather) -----------------
__global__ void pack_b_kernel(const float* __restrict__ act,
                              _Float16* __restrict__ patches,
                              int K, int Ntot) {
  int nl = blockIdx.x * blockDim.x + threadIdx.x;
  int ck = blockIdx.y;
  float v = (ck < K) ? act[(size_t)ck * Ntot + nl] : 0.f;
  int kb = ck >> 5, kr = ck & 31;
  int nb = nl >> 4, nr = nl & 15;
  int lane, e; fragPos(kr, nr, &lane, &e);
  int NB = Ntot >> 4;
  patches[((size_t)(kb * NB + nb) * 32 + lane) * 16 + e] = (_Float16)v;
}

// --- WMMA GEMM -------------------------------------------------------------
// 4 waves/block; wave = 64(M) x 16(N). A panel (4 m-tiles x KB k-steps, contiguous
// in fragment layout) is DMA'd into LDS once per block by the TDM (wave 0), then
// read via ds_load_b128. B fragments are software-pipelined from global memory.
__global__ __launch_bounds__(128) void wmma_gemm_kernel(
    const _Float16* __restrict__ Af, const _Float16* __restrict__ Bf,
    float* __restrict__ out, int KB, int nStart, int nLen, int Ntot,
    int M, const float* __restrict__ bias, int relu) {
  extern __shared__ v16h smemA[];            // KB*4096 bytes: A panel for this block row
  int lane   = threadIdx.x & 31;
  int wid    = threadIdx.x >> 5;
  int nTile  = blockIdx.x * 4 + wid;
  int NB     = nLen >> 4;
  int mbBase = blockIdx.y * 4;
  const int panelBytes = KB << 12;           // 4 m-tiles * KB * 1024B
  const _Float16* Asrc = Af + ((size_t)(mbBase * KB) << 9);   // *512 f16

#if defined(__gfx1250__) && __has_builtin(__builtin_amdgcn_tensor_load_to_lds) && __has_builtin(__builtin_amdgcn_s_wait_tensorcnt)
  if (wid == 0) {                            // uniform branch: one TDM op per block
    unsigned ldsOff = (unsigned)(size_t)(void*)smemA;
    unsigned long long ga = (unsigned long long)(const void*)Asrc;
    unsigned n4 = (unsigned)(panelBytes >> 2);   // 4-byte elements, <= 36864
    // D# group0: count=1 | lds_addr | global_addr[56:0] | type=2 (ISA 8.3)
    v4u g0 = { 1u,
               ldsOff,
               (unsigned)(ga & 0xFFFFFFFFu),
               (unsigned)((ga >> 32) & 0x01FFFFFFu) | (2u << 30) };
    // D# group1: data_size=4B; 1-D tensor/tile of n4 elems (ISA 8.4)
    v8i g1 = { (int)(2u << 16),                       // data_size=2 (4B)
               (int)((n4 & 0xFFFFu) << 16),           // tensor_dim0[15:0]
               (int)((n4 >> 16) | (1u << 16)),        // tensor_dim0[31:16] | tensor_dim1=1
               (int)((n4 & 0xFFFFu) << 16),           // tile_dim0 = n4
               0,                                     // tile_dim1=0, tile_dim2=0
               (int)n4,                               // tensor_dim0_stride[31:0]
               0, 0 };
    v4i gz4 = {0, 0, 0, 0};
    v8i gz8 = {0, 0, 0, 0, 0, 0, 0, 0};
    // 6-arg toolchain form: (g0, g1, g2, g3, g?, cpol)
    __builtin_amdgcn_tensor_load_to_lds(g0, g1, gz4, gz4, gz8, 0);
    __builtin_amdgcn_s_wait_tensorcnt(0);
  }
  __syncthreads();
#else
  {
    const uint4* src = (const uint4*)Asrc;
    uint4* dst = (uint4*)smemA;
    int n16 = panelBytes >> 4;
    for (int t = threadIdx.x; t < n16; t += 128) dst[t] = src[t];
    __syncthreads();
  }
#endif

  const v16h* Bp = (const v16h*)Bf;
  v8f z = {0.f, 0.f, 0.f, 0.f, 0.f, 0.f, 0.f, 0.f};
  v8f acc[4] = {z, z, z, z};
  v16h bCur = Bp[(size_t)nTile * 32 + lane];          // kb = 0
  for (int kb = 0; kb < KB; ++kb) {
    int kn = (kb + 1 < KB) ? (kb + 1) : kb;           // pipeline next B fragment
    v16h bNext = Bp[(size_t)(kn * NB + nTile) * 32 + lane];
#pragma unroll
    for (int i = 0; i < 4; ++i) {
      v16h a = smemA[(size_t)(i * KB + kb) * 32 + lane];   // ds_load_b128 x2
      acc[i] = __builtin_amdgcn_wmma_f32_16x16x32_f16(
          false, a, false, bCur, (short)0, acc[i], false, false);
    }
    bCur = bNext;
  }

  int col = nStart + nTile * 16 + (lane & 15);
  int rh  = (lane >> 4) * 8;     // C/D layout: VGPR v -> M = v + 8*(lane>>4)
#pragma unroll
  for (int i = 0; i < 4; ++i) {
    int mBase = (mbBase + i) * 16 + rh;
#pragma unroll
    for (int v = 0; v < 8; ++v) {
      int row = mBase + v;
      if (row < M) {
        float x = acc[i][v];
        if (bias) x += bias[row];
        if (relu) x = fmaxf(x, 0.f);
        out[(size_t)row * Ntot + col] = x;
      }
    }
  }
}

// --- 2x2 maxpool on [C][B*H*W] ---------------------------------------------
__global__ void maxpool_kernel(const float* __restrict__ in, float* __restrict__ out,
                               int C, int Bn, int H, int W) {
  int Ho = H >> 1, Wo = W >> 1;
  int tid = blockIdx.x * blockDim.x + threadIdx.x;
  if (tid >= C * Bn * Ho * Wo) return;
  int wo = tid % Wo; int t = tid / Wo;
  int ho = t % Ho; t /= Ho;
  int b = t % Bn; int c = t / Bn;
  const float* p = in + ((size_t)c * Bn + b) * H * W + (size_t)(2 * ho) * W + 2 * wo;
  float m = fmaxf(fmaxf(p[0], p[1]), fmaxf(p[W], p[W + 1]));
  out[((size_t)c * Bn + b) * Ho * Wo + (size_t)ho * Wo + wo] = m;
}

// --- finalize: [C][B*256] -> [B,C,16,16]; desc gets channel-L2 norm --------
__global__ void finalize_semi_kernel(const float* __restrict__ semi, float* __restrict__ out) {
  const int HW = 256, Cs = 65, Bn = 16;
  int tid = blockIdx.x * blockDim.x + threadIdx.x;
  if (tid >= Bn * Cs * HW) return;
  int p = tid % HW; int t = tid / HW; int m = t % Cs; int b = t / Cs;
  out[tid] = semi[(size_t)m * (Bn * HW) + (size_t)b * HW + p];
}
__global__ void finalize_desc_kernel(const float* __restrict__ desc, float* __restrict__ out) {
  const int HW = 256, Cd = 256, Bn = 16;
  int tid = blockIdx.x * blockDim.x + threadIdx.x;
  if (tid >= Bn * HW) return;
  int b = tid / HW, p = tid % HW;
  float s = 0.f;
  for (int m = 0; m < Cd; ++m) {
    float v = desc[(size_t)m * (Bn * HW) + (size_t)b * HW + p];
    s += v * v;
  }
  float rn = rsqrtf(s);
  for (int m = 0; m < Cd; ++m) {
    float v = desc[(size_t)m * (Bn * HW) + (size_t)b * HW + p];
    out[((size_t)b * Cd + m) * HW + p] = v * rn;
  }
}

// ---------------------------------------------------------------------------
// Host orchestration
// ---------------------------------------------------------------------------
static int ilog2(int v) { int s = 0; while ((1 << s) < v) ++s; return s; }

static void pack_w(hipStream_t s, const float* w, _Float16* frag,
                   int M, int CK, int Mpad, int Kpad) {
  int total = (Mpad / 16) * (Kpad / 32) * 512;
  pack_weights_kernel<<<dim3((total + 255) / 256), 256, 0, s>>>(w, frag, M, CK, Kpad >> 5, total);
}

static void run_deform(hipStream_t s, const float* actIn, int Cin, const Tap* tab,
                       const _Float16* wf, int M, int Mpad, float* actOut,
                       int H, int W, const float* bias, int relu, _Float16* patches) {
  const int Bn = 16;
  int HW = H * W, Ntot = Bn * HW;
  int hwShift = ilog2(HW);
  int CK = Cin * 9;
  int Kpad = (CK + 31) & ~31;
  int KB = Kpad >> 5;
  long long cap = PATCH_ELEMS / Kpad;
  int chunk = (int)(cap & ~255LL);
  if (chunk > Ntot) chunk = Ntot;
  for (int nStart = 0; nStart < Ntot; nStart += chunk) {
    int nLen = (Ntot - nStart < chunk) ? (Ntot - nStart) : chunk;
    gather_patches_kernel<<<dim3(nLen / 256, Kpad), 256, 0, s>>>(
        actIn, tab, patches, CK, nStart, nLen, Ntot, hwShift);
    wmma_gemm_kernel<<<dim3(nLen / 64, Mpad / 64), 128, (size_t)KB << 12, s>>>(
        wf, patches, actOut, KB, nStart, nLen, Ntot, M, bias, relu);
  }
}

static void run_conv1x1(hipStream_t s, const float* actIn, int K, const _Float16* wf,
                        int M, int Mpad, float* out, const float* bias, int relu,
                        _Float16* patches, int Ntot) {
  int KB = K >> 5;   // K = 256 here, multiple of 32
  pack_b_kernel<<<dim3(Ntot / 256, K), 256, 0, s>>>(actIn, patches, K, Ntot);
  wmma_gemm_kernel<<<dim3(Ntot / 64, Mpad / 64), 128, (size_t)KB << 12, s>>>(
      wf, patches, out, KB, 0, Ntot, Ntot, M, bias, relu);
}

extern "C" void kernel_launch(void* const* d_in, const int* in_sizes, int n_in,
                              void* d_out, int out_size, void* d_ws, size_t ws_size,
                              hipStream_t stream) {
  (void)in_sizes; (void)n_in; (void)out_size; (void)ws_size;
  const float* x    = (const float*)d_in[0];
  const float* w1a  = (const float*)d_in[1];
  const float* w1b  = (const float*)d_in[2];
  const float* w2a  = (const float*)d_in[3];
  const float* w2b  = (const float*)d_in[4];
  const float* w3a  = (const float*)d_in[5];
  const float* w3b  = (const float*)d_in[6];
  const float* w4a  = (const float*)d_in[7];
  const float* w4b  = (const float*)d_in[8];
  const float* wPa  = (const float*)d_in[9];
  const float* wPb  = (const float*)d_in[10];
  const float* bPb  = (const float*)d_in[11];
  const float* wDa  = (const float*)d_in[12];
  const float* wDb  = (const float*)d_in[13];
  const float* bDb  = (const float*)d_in[14];
  const float* c128 = (const float*)d_in[15];
  const float* c64  = (const float*)d_in[16];
  const float* c32  = (const float*)d_in[17];
  const float* c16  = (const float*)d_in[18];

  char* ws = (char*)d_ws;
  size_t off = 0;
  auto alloc = [&](size_t bytes) -> void* {
    void* p = ws + off;
    off = (off + bytes + 255) & ~(size_t)255;
    return p;
  };

  Tap* tab128 = (Tap*)alloc(9ull * 16384 * sizeof(Tap));
  Tap* tab64  = (Tap*)alloc(9ull * 4096  * sizeof(Tap));
  Tap* tab32  = (Tap*)alloc(9ull * 1024  * sizeof(Tap));
  Tap* tab16  = (Tap*)alloc(9ull * 256   * sizeof(Tap));
  _Float16* f1a = (_Float16*)alloc(2ull * 64  * 32);
  _Float16* f1b = (_Float16*)alloc(2ull * 64  * 576);
  _Float16* f2a = (_Float16*)alloc(2ull * 64  * 576);
  _Float16* f2b = (_Float16*)alloc(2ull * 64  * 576);
  _Float16* f3a = (_Float16*)alloc(2ull * 128 * 576);
  _Float16* f3b = (_Float16*)alloc(2ull * 128 * 1152);
  _Float16* f4a = (_Float16*)alloc(2ull * 128 * 1152);
  _Float16* f4b = (_Float16*)alloc(2ull * 128 * 1152);
  _Float16* fPa = (_Float16*)alloc(2ull * 256 * 1152);
  _Float16* fPb = (_Float16*)alloc(2ull * 128 * 256);
  _Float16* fDa = (_Float16*)alloc(2ull * 256 * 1152);
  _Float16* fDb = (_Float16*)alloc(2ull * 256 * 256);
  float* bufA = (float*)alloc(4ull * 64 * 262144);
  float* bufB = (float*)alloc(4ull * 64 * 262144);
  float* bufC = (float*)alloc(4ull * 65 * 4096);
  _Float16* patches = (_Float16*)alloc(2ull * (size_t)PATCH_ELEMS);

  // bilinear tap tables (batch-invariant)
  build_tables_kernel<<<dim3((9 * 16384 + 255) / 256), 256, 0, stream>>>(c128, tab128, 128, 128);
  build_tables_kernel<<<dim3((9 * 4096  + 255) / 256), 256, 0, stream>>>(c64,  tab64,  64,  64);
  build_tables_kernel<<<dim3((9 * 1024  + 255) / 256), 256, 0, stream>>>(c32,  tab32,  32,  32);
  build_tables_kernel<<<dim3((9 * 256   + 255) / 256), 256, 0, stream>>>(c16,  tab16,  16,  16);

  // weight fragments
  pack_w(stream, w1a, f1a, 64,  9,    64,  32);
  pack_w(stream, w1b, f1b, 64,  576,  64,  576);
  pack_w(stream, w2a, f2a, 64,  576,  64,  576);
  pack_w(stream, w2b, f2b, 64,  576,  64,  576);
  pack_w(stream, w3a, f3a, 128, 576,  128, 576);
  pack_w(stream, w3b, f3b, 128, 1152, 128, 1152);
  pack_w(stream, w4a, f4a, 128, 1152, 128, 1152);
  pack_w(stream, w4b, f4b, 128, 1152, 128, 1152);
  pack_w(stream, wPa, fPa, 256, 1152, 256, 1152);
  pack_w(stream, wPb, fPb, 65,  256,  128, 256);
  pack_w(stream, wDa, fDa, 256, 1152, 256, 1152);
  pack_w(stream, wDb, fDb, 256, 256,  256, 256);

  const int Bn = 16;
  // stage 1 @128x128
  run_deform(stream, x,    1,  tab128, f1a, 64, 64, bufA, 128, 128, nullptr, 1, patches);
  run_deform(stream, bufA, 64, tab128, f1b, 64, 64, bufB, 128, 128, nullptr, 1, patches);
  maxpool_kernel<<<dim3((64 * Bn * 64 * 64 + 255) / 256), 256, 0, stream>>>(bufB, bufA, 64, Bn, 128, 128);
  // stage 2 @64x64
  run_deform(stream, bufA, 64, tab64, f2a, 64, 64, bufB, 64, 64, nullptr, 1, patches);
  run_deform(stream, bufB, 64, tab64, f2b, 64, 64, bufA, 64, 64, nullptr, 1, patches);
  maxpool_kernel<<<dim3((64 * Bn * 32 * 32 + 255) / 256), 256, 0, stream>>>(bufA, bufB, 64, Bn, 64, 64);
  // stage 3 @32x32
  run_deform(stream, bufB, 64,  tab32, f3a, 128, 128, bufA, 32, 32, nullptr, 1, patches);
  run_deform(stream, bufA, 128, tab32, f3b, 128, 128, bufB, 32, 32, nullptr, 1, patches);
  maxpool_kernel<<<dim3((128 * Bn * 16 * 16 + 255) / 256), 256, 0, stream>>>(bufB, bufA, 128, Bn, 32, 32);
  // stage 4 @16x16
  run_deform(stream, bufA, 128, tab16, f4a, 128, 128, bufB, 16, 16, nullptr, 1, patches);
  run_deform(stream, bufB, 128, tab16, f4b, 128, 128, bufA, 16, 16, nullptr, 1, patches);  // h -> A
  // detector head
  run_deform(stream, bufA, 128, tab16, fPa, 256, 256, bufB, 16, 16, nullptr, 1, patches);  // cPa -> B
  run_conv1x1(stream, bufB, 256, fPb, 65, 128, bufC, bPb, 0, patches, Bn * 256);           // semi -> C
  // descriptor head
  run_deform(stream, bufA, 128, tab16, fDa, 256, 256, bufB, 16, 16, nullptr, 1, patches);  // cDa -> B
  run_conv1x1(stream, bufB, 256, fDb, 256, 256, bufA, bDb, 0, patches, Bn * 256);          // desc -> A

  float* outF = (float*)d_out;
  finalize_semi_kernel<<<dim3((Bn * 65 * 256 + 255) / 256), 256, 0, stream>>>(bufC, outF);
  finalize_desc_kernel<<<dim3((Bn * 256 + 255) / 256), 256, 0, stream>>>(bufA, outF + (size_t)Bn * 65 * 256);
}
